// StereoDepthEstimationModel_51118700757039
// MI455X (gfx1250) — matrix operand order repaired and checked by reading, not compiled, not run on hardware
//
#include <hip/hip_runtime.h>
#include <hip/hip_bf16.h>

typedef _Float16 h2   __attribute__((ext_vector_type(2)));
typedef _Float16 v8h  __attribute__((ext_vector_type(8)));
typedef _Float16 v16h __attribute__((ext_vector_type(16)));
typedef float    v8f  __attribute__((ext_vector_type(8)));

static constexpr int kH  = 384;
static constexpr int kW  = 1248;
static constexpr int kHP = kH + 4;   // padded (2-halo) height
static constexpr int kWP = kW + 4;   // padded width
static constexpr int kC  = 32;
static constexpr int kD  = 64;       // MAX_DISPARITY

// ---------------------------------------------------------------------------
// Kernel 1: transpose w2 [oc][ic][ky][kx] f32 -> wtap [tap][oc][ic] f16
// (ic contiguous so WMMA A-fragment gathers are two 16B loads)
// ---------------------------------------------------------------------------
__global__ __launch_bounds__(256) void sdem_prep_w2(const float* __restrict__ w2,
                                                    _Float16* __restrict__ wtap) {
  int idx = blockIdx.x * 256 + threadIdx.x;
  if (idx >= 25 * 32 * 32) return;
  int t  = idx / 1024;
  int oc = (idx / 32) % 32;
  int ic = idx % 32;
  wtap[idx] = (_Float16)w2[(oc * 32 + ic) * 25 + t];
}

// ---------------------------------------------------------------------------
// Kernel 2: conv1 (1->32ch, 5x5, SAME) + bias + relu for both images.
// Output: padded channel-last f16 [img][kHP][kWP][32] (halo pre-zeroed).
// ---------------------------------------------------------------------------
__global__ __launch_bounds__(256) void sdem_conv1(const float* __restrict__ left,
                                                  const float* __restrict__ right,
                                                  const float* __restrict__ w1,
                                                  const float* __restrict__ b1,
                                                  _Float16* __restrict__ f1) {
  int idx = blockIdx.x * 256 + threadIdx.x;
  if (idx >= 2 * kH * kW) return;
  int img = idx / (kH * kW);
  int rem = idx % (kH * kW);
  int h = rem / kW, w = rem % kW;
  const float* src = img ? right : left;

  float xin[25];
#pragma unroll
  for (int ky = 0; ky < 5; ++ky)
#pragma unroll
    for (int kx = 0; kx < 5; ++kx) {
      int hh = h + ky - 2, ww = w + kx - 2;
      xin[ky * 5 + kx] = (hh >= 0 && hh < kH && ww >= 0 && ww < kW) ? src[hh * kW + ww] : 0.0f;
    }

  _Float16 o[32];
#pragma unroll
  for (int oc = 0; oc < 32; ++oc) {
    float acc = b1[oc];
#pragma unroll
    for (int t = 0; t < 25; ++t) acc = fmaf(w1[oc * 25 + t], xin[t], acc);
    o[oc] = (_Float16)fmaxf(acc, 0.0f);
  }
  _Float16* dst = f1 + ((size_t)img * kHP * kWP + (size_t)(h + 2) * kWP + (w + 2)) * kC;
  const uint4* so = (const uint4*)o;
  uint4* gd = (uint4*)dst;
  gd[0] = so[0]; gd[1] = so[1]; gd[2] = so[2]; gd[3] = so[3];
}

// ---------------------------------------------------------------------------
// Kernel 3: conv2 (32->32ch, 5x5) as 25 per-tap GEMMs via v_wmma_f32_16x16x32_f16.
// Block = 64 threads (2 waves); wave m handles output channels m*16..m*16+15 of a
// 16-pixel tile on one row. K = 32 input channels per tap.
// ---------------------------------------------------------------------------
__global__ __launch_bounds__(64) void sdem_conv2_wmma(const _Float16* __restrict__ f1,
                                                      const _Float16* __restrict__ wtap,
                                                      const float* __restrict__ b2,
                                                      _Float16* __restrict__ f2) {
  __shared__ __align__(64) _Float16 smem[5 * 20 * 32];  // 5 rows x 20 px x 32 ch
  const int tid  = threadIdx.x;
  const int lane = tid & 31;
  const int mb   = tid >> 5;          // wave id == M-block (output-channel block)
  const int w0   = blockIdx.x * 16;   // tile base (same index in padded coords)
  const int h    = blockIdx.y;
  const int img  = blockIdx.z;
  const _Float16* f1img = f1 + (size_t)img * kHP * kWP * kC;

  __builtin_prefetch(f1img + ((size_t)h * kWP + w0) * kC, 0, 1);  // global_prefetch_b8

  // Cooperative stage: rows h..h+4 (padded), cols w0..w0+19, all 32 channels.
  for (int c = tid; c < (5 * 20 * 32) / 8; c += 64) {
    int idx = c * 8;
    int r   = idx / (20 * 32);
    int rm  = idx % (20 * 32);
    int px  = rm / 32;
    int ch  = rm % 32;
    *(uint4*)(smem + idx) =
        *(const uint4*)(f1img + ((size_t)(h + r) * kWP + (w0 + px)) * kC + ch);
  }
  __syncthreads();

  const int n = lane & 15;   // pixel column within tile (B/D fragment N)
  const int g = lane >> 4;   // half-wave select
  v8f acc = {};
#pragma unroll
  for (int t = 0; t < 25; ++t) {
    const int dy = t / 5, dx = t % 5;
    // A fragment (16x32 f16): row M = mb*16+n; lanes 0-15 hold K 0-15,
    // lanes 16-31 hold K 16-31 (contiguous, matching B).
    const _Float16* wp = wtap + ((size_t)(t * 32) + mb * 16 + n) * 32;
    v8h alo = *(const v8h*)(wp + g * 16);
    v8h ahi = *(const v8h*)(wp + g * 16 + 8);
    v16h a = __builtin_shufflevector(alo, ahi, 0, 1, 2, 3, 4, 5, 6, 7,
                                     8, 9, 10, 11, 12, 13, 14, 15);
    // B fragment (32x16 f16): column N = n; 16 contiguous K values per lane.
    v16h b = *(const v16h*)(smem + ((dy * 20 + dx + n) * 32 + g * 16));
    acc = __builtin_amdgcn_wmma_f32_16x16x32_f16(false, a, false, b,
                                                 (short)0, acc, false, false);
  }
  __syncthreads();

  // Repack D (M = r + 8*g per ISA C/D layout) + bias + relu into LDS [px][ch].
#pragma unroll
  for (int r = 0; r < 8; ++r) {
    int oc = mb * 16 + g * 8 + r;
    smem[n * 32 + oc] = (_Float16)fmaxf(acc[r] + b2[oc], 0.0f);
  }
  __syncthreads();

  // Coalesced 128-bit write-out: 64 threads x 16B = 16 px x 32 ch.
  {
    int idx = tid * 8;
    int px = idx / 32, ch = idx % 32;
    _Float16* dst = f2 + ((size_t)img * kH * kW + (size_t)h * kW + (w0 + px)) * kC + ch;
    *(uint4*)dst = *(const uint4*)(smem + idx);
  }
}

// ---------------------------------------------------------------------------
// Kernel 4: fused SAD cost + argmin over disparity per (h,w).
// Packed-f16 math: v_pk_add_f16 (sub via neg) + sign-mask AND for |.|.
// Right features staged in LDS with a 63-pixel left halo (zero-filled OOB,
// matching the reference's zero padding of the shifted right features).
// ---------------------------------------------------------------------------
__global__ __launch_bounds__(256) void sdem_sad(const _Float16* __restrict__ f2,
                                                float* __restrict__ minval,
                                                int* __restrict__ argd) {
  __shared__ _Float16 sfl[256 * 32];
  __shared__ _Float16 sfr[320 * 32];
  const int tid = threadIdx.x;
  const int h   = blockIdx.y;
  const int w0  = blockIdx.x * 256;
  const _Float16* fl = f2 + (size_t)h * kW * kC;
  const _Float16* fr = f2 + ((size_t)kH * kW + (size_t)h * kW) * kC;

  for (int c = tid; c < 256 * 32 / 8; c += 256) {
    int idx = c * 8;
    int px = idx / 32, ch = idx % 32;
    int gw = w0 + px;
    uint4 v = {0, 0, 0, 0};
    if (gw < kW) v = *(const uint4*)(fl + (size_t)gw * kC + ch);
    *(uint4*)(sfl + idx) = v;
  }
  for (int c = tid; c < 320 * 32 / 8; c += 256) {
    int idx = c * 8;
    int px = idx / 32, ch = idx % 32;
    int gw = w0 - 63 + px;
    uint4 v = {0, 0, 0, 0};
    if (gw >= 0 && gw < kW) v = *(const uint4*)(fr + (size_t)gw * kC + ch);
    *(uint4*)(sfr + idx) = v;
  }
  __syncthreads();

  const int w = w0 + tid;
  if (w >= kW) return;

  h2 flv[16];
  const h2* flp = (const h2*)(sfl + tid * 32);
#pragma unroll
  for (int j = 0; j < 16; ++j) flv[j] = flp[j];

  float best = 3.4e38f;
  int bestd = 0;
  for (int d = 0; d < kD; ++d) {
    const h2* frp = (const h2*)(sfr + (size_t)(tid + 63 - d) * 32);
    h2 sacc = {(_Float16)0, (_Float16)0};
#pragma unroll
    for (int j = 0; j < 16; ++j) {
      h2 t = flv[j] - frp[j];
      unsigned u = __builtin_bit_cast(unsigned, t) & 0x7FFF7FFFu;
      sacc = sacc + __builtin_bit_cast(h2, u);
    }
    float cost = ((float)sacc[0] + (float)sacc[1]) * (1.0f / 32.0f);
    if (cost < best) { best = cost; bestd = d; }  // strict '<' keeps first d
  }
  minval[(size_t)h * kW + w] = best;
  argd[(size_t)h * kW + w]   = bestd;
}

// ---------------------------------------------------------------------------
// Kernel 5: per-column argmin over h; flattened index = d*kH + h.
// ---------------------------------------------------------------------------
__global__ __launch_bounds__(256) void sdem_argmin_col(const float* __restrict__ minval,
                                                       const int* __restrict__ argd,
                                                       int* __restrict__ out) {
  int w = blockIdx.x * 256 + threadIdx.x;
  if (w >= kW) return;
  float best = 3.4e38f;
  int bi = 0;
  for (int hh = 0; hh < kH; ++hh) {
    float v = minval[(size_t)hh * kW + w];
    if (v < best) { best = v; bi = argd[(size_t)hh * kW + w] * kH + hh; }
  }
  out[w] = bi;
}

// ---------------------------------------------------------------------------
extern "C" void kernel_launch(void* const* d_in, const int* in_sizes, int n_in,
                              void* d_out, int out_size, void* d_ws, size_t ws_size,
                              hipStream_t stream) {
  (void)in_sizes; (void)n_in; (void)out_size;
  const float* left  = (const float*)d_in[0];
  const float* right = (const float*)d_in[1];
  const float* w1    = (const float*)d_in[2];
  const float* b1    = (const float*)d_in[3];
  const float* w2    = (const float*)d_in[4];
  const float* b2    = (const float*)d_in[5];
  int* out = (int*)d_out;

  const size_t SZ_WTAP = (size_t)25 * 32 * 32 * sizeof(_Float16);         // 51200
  const size_t OFF_F1  = SZ_WTAP;                                         // 16B-aligned
  const size_t SZ_F1   = (size_t)2 * kHP * kWP * kC * sizeof(_Float16);
  const size_t OFF_F2  = OFF_F1 + SZ_F1;
  const size_t SZ_F2   = (size_t)2 * kH * kW * kC * sizeof(_Float16);
  const size_t OFF_MIN = OFF_F2 + SZ_F2;
  const size_t SZ_MIN  = (size_t)kH * kW * sizeof(float);
  const size_t OFF_ARG = OFF_MIN + SZ_MIN;
  const size_t SZ_ARG  = (size_t)kH * kW * sizeof(int);
  if (ws_size < OFF_ARG + SZ_ARG) return;

  char* ws = (char*)d_ws;
  _Float16* wtap  = (_Float16*)(ws);
  _Float16* f1    = (_Float16*)(ws + OFF_F1);
  _Float16* f2    = (_Float16*)(ws + OFF_F2);
  float*    minv  = (float*)(ws + OFF_MIN);
  int*      argd  = (int*)(ws + OFF_ARG);

  hipMemsetAsync(f1, 0, SZ_F1, stream);  // zero padded halo (graph-capturable)

  sdem_prep_w2<<<(25 * 32 * 32 + 255) / 256, 256, 0, stream>>>(w2, wtap);
  sdem_conv1<<<(2 * kH * kW + 255) / 256, 256, 0, stream>>>(left, right, w1, b1, f1);
  sdem_conv2_wmma<<<dim3(kW / 16, kH, 2), 64, 0, stream>>>(f1, wtap, b2, f2);
  sdem_sad<<<dim3((kW + 255) / 256, kH), 256, 0, stream>>>(f2, minv, argd);
  sdem_argmin_col<<<(kW + 255) / 256, 256, 0, stream>>>(minv, argd, out);
}
